// DeformableAttention_56307021250859
// MI455X (gfx1250) — compile-verified, matrix operand order checked by reading, not run on hardware
//
#include <hip/hip_runtime.h>
#include <hip/hip_bf16.h>
#include <hip/hip_fp16.h>

// ---------------------------------------------------------------------------
// DeformableAttention for MI455X (gfx1250, wave32, WMMA + async-to-LDS)
//   B=16, H=W=64, C=256, HEADS=8, DIM_HEAD=32, INNER=256, N=4096 tokens/batch
// ---------------------------------------------------------------------------

typedef __attribute__((ext_vector_type(16))) _Float16 v16h;
typedef __attribute__((ext_vector_type(8)))  float    v8f;

#if defined(__gfx1250__) && __has_builtin(__builtin_amdgcn_global_load_async_to_lds_b128)
#define USE_ASYNC_LDS 1
#else
#define USE_ASYNC_LDS 0
#endif

#if USE_ASYNC_LDS
// The builtin's params are: (int4 __device__ *src, int4 __shared__ *dst, imm, imm)
typedef int v4i_b128 __attribute__((vector_size(16)));
typedef __attribute__((address_space(1))) v4i_b128 gv4i_t;
typedef __attribute__((address_space(3))) v4i_b128 sv4i_t;
// generic->global: 64-bit identity; generic->LDS: low 32 bits are the LDS
// byte offset (aperture lives in the high bits per CDNA5 ISA 10.2).
__device__ __forceinline__ gv4i_t* as_global(const void* p) {
    return (gv4i_t*)(unsigned long long)p;
}
__device__ __forceinline__ sv4i_t* as_shared(void* p) {
    return (sv4i_t*)(unsigned int)(unsigned long long)p;
}
__device__ __forceinline__ void wait_async_le6() {
#if __has_builtin(__builtin_amdgcn_s_wait_asynccnt)
    __builtin_amdgcn_s_wait_asynccnt(6);
#else
    asm volatile("s_wait_asynccnt 0x6" ::: "memory");
#endif
}
__device__ __forceinline__ void wait_async_le0() {
#if __has_builtin(__builtin_amdgcn_s_wait_asynccnt)
    __builtin_amdgcn_s_wait_asynccnt(0);
#else
    asm volatile("s_wait_asynccnt 0x0" ::: "memory");
#endif
}
#endif

// ---------------------------------------------------------------------------
// Kernel 1: fused QKV GEMM  [65536 x 256] x [256 x 768] -> q/k/v (f16)
// Block: 256 threads = 8 waves; tile 128 rows x 64 cols; wave tile 32x32.
// Double-buffered LDS staging via async global->LDS copies (ASYNCcnt).
// Output layouts (all f16):  buf[((b*8+head)*4096 + tok)*32 + dch]
// ---------------------------------------------------------------------------

// Stage one K-tile (A: 128x32 f32, B: 32x64 f32) into LDS buffer `buf`.
// Each thread moves 4 A-float4s + 2 B-float4s (6 x b128 per lane).
__device__ __forceinline__ void stage_tile(
    const float* __restrict__ x, const float* __restrict__ wsrc,
    int rowBase, int wstride, int wcol0, int kb,
    float* __restrict__ As, float* __restrict__ Bs, int tid)
{
#if USE_ASYNC_LDS
    #pragma unroll
    for (int i = 0; i < 4; ++i) {
        int chunk = tid + i * 256;
        int r = chunk >> 3, c4 = chunk & 7;
        const float* g = x + (size_t)(rowBase + r) * 256 + kb * 32 + c4 * 4;
        __builtin_amdgcn_global_load_async_to_lds_b128(
            as_global(g), as_shared(As + r * 32 + c4 * 4), 0, 0);
    }
    #pragma unroll
    for (int i = 0; i < 2; ++i) {
        int chunk = tid + i * 256;
        int r = chunk >> 4, c4 = chunk & 15;
        const float* g = wsrc + (size_t)(kb * 32 + r) * wstride + wcol0 + c4 * 4;
        __builtin_amdgcn_global_load_async_to_lds_b128(
            as_global(g), as_shared(Bs + r * 64 + c4 * 4), 0, 0);
    }
#else
    // fallback: batch all global loads into regs first so they pipeline,
    // then drain to LDS.
    float4 va[4], vb[2];
    int ra[4], ca[4], rb[2], cb2[2];
    #pragma unroll
    for (int i = 0; i < 4; ++i) {
        int chunk = tid + i * 256;
        ra[i] = chunk >> 3; ca[i] = chunk & 7;
        va[i] = *(const float4*)(x + (size_t)(rowBase + ra[i]) * 256 + kb * 32 + ca[i] * 4);
    }
    #pragma unroll
    for (int i = 0; i < 2; ++i) {
        int chunk = tid + i * 256;
        rb[i] = chunk >> 4; cb2[i] = chunk & 15;
        vb[i] = *(const float4*)(wsrc + (size_t)(kb * 32 + rb[i]) * wstride + wcol0 + cb2[i] * 4);
    }
    #pragma unroll
    for (int i = 0; i < 4; ++i) *(float4*)(&As[ra[i] * 32 + ca[i] * 4]) = va[i];
    #pragma unroll
    for (int i = 0; i < 2; ++i) *(float4*)(&Bs[rb[i] * 64 + cb2[i] * 4]) = vb[i];
#endif
}

__global__ __launch_bounds__(256) void qkv_gemm(
    const float* __restrict__ x,        // [65536][256]
    const float* __restrict__ Wq,       // [256][256]
    const float* __restrict__ Wkv,      // [256][512]
    _Float16* __restrict__ qbuf,
    _Float16* __restrict__ kbuf,
    _Float16* __restrict__ vbuf)
{
    __shared__ float As[2][128 * 32];   // 2 x 16 KB
    __shared__ float Bs[2][32 * 64];    // 2 x  8 KB

    const int tid   = threadIdx.x;
    const int lane  = tid & 31;
    const int wid   = tid >> 5;         // 0..7
    const int waveM = wid & 3;          // 4 strips of 32 rows
    const int waveN = wid >> 2;         // 2 strips of 32 cols
    const int rowBase = blockIdx.x * 128;
    const int nBase   = blockIdx.y * 64;

    const float* wsrc;
    int wstride, wcol0;
    if (nBase < 256) { wsrc = Wq;  wstride = 256; wcol0 = nBase; }
    else             { wsrc = Wkv; wstride = 512; wcol0 = nBase - 256; }

    v8f acc[2][2] = {};

    const int half = lane >> 4;         // 0 or 1
    const int kb8  = half * 8;          // K sub-offset for this lane half

    // prologue: stage tile 0 into buffer 0
    stage_tile(x, wsrc, rowBase, wstride, wcol0, 0, As[0], Bs[0], tid);

    for (int kb = 0; kb < 8; ++kb) {
        const int cur = kb & 1;
        if (kb < 7) {
            // issue next tile's async copies into the other buffer, then wait
            // only for the *current* tile (per-wave ASYNCcnt is in-order:
            // <=6 outstanding means the previous 6 have landed).
            stage_tile(x, wsrc, rowBase, wstride, wcol0, kb + 1,
                       As[cur ^ 1], Bs[cur ^ 1], tid);
#if USE_ASYNC_LDS
            wait_async_le6();
#endif
        } else {
#if USE_ASYNC_LDS
            wait_async_le0();
#endif
        }
        __syncthreads();

        // -------- build fragments (16-bit A/B layout per CDNA5 ISA 7.12.2) --
        const float* __restrict__ Ab = As[cur];
        const float* __restrict__ Bb = Bs[cur];
        v16h afr[2], bfr[2];
        #pragma unroll
        for (int s = 0; s < 2; ++s) {
            int row = waveM * 32 + s * 16 + (lane & 15);
            const float* p0 = &Ab[row * 32 + kb8];        // K = kb8..kb8+7
            const float* p1 = &Ab[row * 32 + 16 + kb8];   // K = 16+kb8..
            #pragma unroll
            for (int j = 0; j < 8; ++j) {
                afr[s][j]     = (_Float16)p0[j];
                afr[s][8 + j] = (_Float16)p1[j];
            }
        }
        #pragma unroll
        for (int s = 0; s < 2; ++s) {
            int col = waveN * 32 + s * 16 + (lane & 15);
            #pragma unroll
            for (int j = 0; j < 8; ++j) {
                bfr[s][j]     = (_Float16)Bb[(kb8 + j) * 64 + col];
                bfr[s][8 + j] = (_Float16)Bb[(16 + kb8 + j) * 64 + col];
            }
        }
        #pragma unroll
        for (int i = 0; i < 2; ++i)
            #pragma unroll
            for (int j2 = 0; j2 < 2; ++j2)
                acc[i][j2] = __builtin_amdgcn_wmma_f32_16x16x32_f16(
                    false, afr[i], false, bfr[j2], (short)0, acc[i][j2], false, false);
        __syncthreads();
    }

    // -------- scatter C tiles to q/k/v f16 layouts --------
    const int reg = nBase >> 8;   // 0:q 1:k 2:v (64-wide tile never straddles)
    _Float16* dst = (reg == 0) ? qbuf : (reg == 1) ? kbuf : vbuf;
    #pragma unroll
    for (int i = 0; i < 2; ++i)
        #pragma unroll
        for (int j2 = 0; j2 < 2; ++j2)
            #pragma unroll
            for (int r = 0; r < 8; ++r) {
                int M = r + half * 8;               // C layout: vgpr r, lane half
                int N = lane & 15;
                int gtok = rowBase + waveM * 32 + i * 16 + M;
                int gn   = nBase + waveN * 32 + j2 * 16 + N;
                int b = gtok >> 12, tok = gtok & 4095;
                int cn = gn & 255;
                int head = cn >> 5, dch = cn & 31;
                size_t di = ((size_t)(b * 8 + head) * 4096 + tok) * 32 + dch;
                dst[di] = (_Float16)acc[i][j2][r];
            }
}

// ---------------------------------------------------------------------------
// Kernel 2: 3x3 offset conv as implicit-im2col WMMA GEMM.
// M = pixels (wave tile 16), N = 16 out channels, K = 9*256 = 2304 (tap-major)
// Block: 128 threads = 4 waves = 64 pixels.
// off layout: off[(b*16 + oc)*4096 + pix]
// ---------------------------------------------------------------------------
__global__ __launch_bounds__(128) void offset_conv(
    const float* __restrict__ x_in,     // [16][64][64][256]
    const float* __restrict__ cw,       // [16][256][3][3]
    const float* __restrict__ cb,       // [16]
    float* __restrict__ off)
{
    const int tid  = threadIdx.x;
    const int lane = tid & 31;
    const int wid  = tid >> 5;
    const int pixBase = blockIdx.x * 64 + wid * 16;   // global pixel (b*4096+pix)
    const int b = pixBase >> 12;
    const int half = lane >> 4;
    const int kb8  = half * 8;

    v8f acc = {};

    for (int kb = 0; kb < 72; ++kb) {
        const int tap  = kb >> 3;            // 0..8  (K block never crosses taps)
        const int coff = (kb & 7) * 32;      // channel offset within tap
        const int dy = tap / 3 - 1, dx = tap % 3 - 1;

        // A fragment: 16 pixels x 32 channels of the shifted image (zero pad)
        const int prow = pixBase + (lane & 15);
        const int pix  = prow & 4095;
        const int yy = pix >> 6, xx = pix & 63;
        const int sy = yy + dy, sx = xx + dx;
        v16h afr;
        if (sy >= 0 && sy < 64 && sx >= 0 && sx < 64) {
            const float* p = x_in + (((size_t)b * 64 + sy) * 64 + sx) * 256 + coff;
            #pragma unroll
            for (int j = 0; j < 8; ++j) {
                afr[j]     = (_Float16)p[kb8 + j];
                afr[8 + j] = (_Float16)p[16 + kb8 + j];
            }
        } else {
            #pragma unroll
            for (int j = 0; j < 16; ++j) afr[j] = (_Float16)0.f;
        }

        // B fragment: weights, col = out channel, k -> (tap, channel)
        const int n = lane & 15;
        const float* wp = cw + (size_t)n * 2304 + tap;  // + c*9 below
        v16h bfr;
        #pragma unroll
        for (int j = 0; j < 8; ++j) {
            bfr[j]     = (_Float16)wp[(size_t)(coff + kb8 + j) * 9];
            bfr[8 + j] = (_Float16)wp[(size_t)(coff + 16 + kb8 + j) * 9];
        }

        acc = __builtin_amdgcn_wmma_f32_16x16x32_f16(
            false, afr, false, bfr, (short)0, acc, false, false);
    }

    const int N = lane & 15;
    const float bias = cb[N];
    #pragma unroll
    for (int r = 0; r < 8; ++r) {
        int M = r + half * 8;
        int pix = (pixBase + M) & 4095;
        off[((size_t)b * 16 + N) * 4096 + pix] = acc[r] + bias;
    }
}

// ---------------------------------------------------------------------------
// Kernel 3: bilinear sampling + per-token logit + softmax over 4096 tokens +
//           weighted v reduction. One workgroup per (b, head).
// ---------------------------------------------------------------------------
struct SampInfo { size_t i00, i01, i10, i11; float w00, w01, w10, w11; };

__device__ __forceinline__ SampInfo samp_coords(const float* __restrict__ off,
                                                int b, int head, int t)
{
    float ox = off[((size_t)b * 16 + 2 * head)     * 4096 + t];
    float oy = off[((size_t)b * 16 + 2 * head + 1) * 4096 + t];
    int ty = t >> 6, tx = t & 63;
    float gx = -1.f + 2.f * tx / 63.f;
    float gy = -1.f + 2.f * ty / 63.f;
    float px = fminf(fmaxf((gx + ox + 1.f) * 0.5f * 63.f, 0.f), 63.f);
    float py = fminf(fmaxf((gy + oy + 1.f) * 0.5f * 63.f, 0.f), 63.f);
    float x0f = floorf(px), y0f = floorf(py);
    float wx = px - x0f, wy = py - y0f;
    int x0 = (int)x0f, y0 = (int)y0f;
    int x1 = min(x0 + 1, 63), y1 = min(y0 + 1, 63);
    SampInfo s;
    s.i00 = ((size_t)y0 * 64 + x0) * 32;  s.i01 = ((size_t)y0 * 64 + x1) * 32;
    s.i10 = ((size_t)y1 * 64 + x0) * 32;  s.i11 = ((size_t)y1 * 64 + x1) * 32;
    s.w00 = (1.f - wy) * (1.f - wx);  s.w01 = (1.f - wy) * wx;
    s.w10 = wy * (1.f - wx);          s.w11 = wy * wx;
    return s;
}

__global__ __launch_bounds__(256) void attn_reduce(
    const _Float16* __restrict__ qbuf,
    const _Float16* __restrict__ kbuf,
    const _Float16* __restrict__ vbuf,
    const float* __restrict__ off,
    float* __restrict__ osmall)         // [16*8][32]
{
    __shared__ float logits[4096];
    __shared__ float red[256];
    __shared__ float vred[256 * 32];

    const int bh = blockIdx.x;
    const int b = bh >> 3, head = bh & 7;
    const int tid = threadIdx.x;
    const size_t base = (size_t)bh * 4096 * 32;

    // ---- phase 1: logits = <q_norm, sampled_k_norm> ----
    for (int t = tid; t < 4096; t += 256) {
        float q[32]; float ss = 0.f;
        #pragma unroll
        for (int j = 0; j < 32; ++j) {
            q[j] = (float)qbuf[base + (size_t)t * 32 + j];
            ss += q[j] * q[j];
        }
        float qinv = 1.f / fmaxf(sqrtf(ss), 1e-12f);

        SampInfo s = samp_coords(off, b, head, t);
        const _Float16* kp = kbuf + base;
        float dot = 0.f, ks = 0.f;
        #pragma unroll
        for (int j = 0; j < 32; ++j) {
            float v = s.w00 * (float)kp[s.i00 + j] + s.w01 * (float)kp[s.i01 + j]
                    + s.w10 * (float)kp[s.i10 + j] + s.w11 * (float)kp[s.i11 + j];
            ks  += v * v;
            dot += q[j] * v;
        }
        float kinv = 1.f / fmaxf(sqrtf(ks), 1e-12f);
        logits[t] = dot * qinv * kinv;
    }
    __syncthreads();

    // ---- softmax statistics over 4096 tokens ----
    float mx = -1e30f;
    for (int t = tid; t < 4096; t += 256) mx = fmaxf(mx, logits[t]);
    red[tid] = mx; __syncthreads();
    for (int st = 128; st > 0; st >>= 1) {
        if (tid < st) red[tid] = fmaxf(red[tid], red[tid + st]);
        __syncthreads();
    }
    mx = red[0]; __syncthreads();

    float sum = 0.f;
    for (int t = tid; t < 4096; t += 256) sum += __expf(logits[t] - mx);
    red[tid] = sum; __syncthreads();
    for (int st = 128; st > 0; st >>= 1) {
        if (tid < st) red[tid] += red[tid + st];
        __syncthreads();
    }
    const float invsum = 1.f / red[0];

    // ---- phase 2: out = sum_t softmax_t * sampled_v(t) ----
    float acc[32];
    #pragma unroll
    for (int j = 0; j < 32; ++j) acc[j] = 0.f;
    for (int t = tid; t < 4096; t += 256) {
        float p = __expf(logits[t] - mx) * invsum;
        SampInfo s = samp_coords(off, b, head, t);
        const _Float16* vp = vbuf + base;
        #pragma unroll
        for (int j = 0; j < 32; ++j) {
            float v = s.w00 * (float)vp[s.i00 + j] + s.w01 * (float)vp[s.i01 + j]
                    + s.w10 * (float)vp[s.i10 + j] + s.w11 * (float)vp[s.i11 + j];
            acc[j] += p * v;
        }
    }
    #pragma unroll
    for (int j = 0; j < 32; ++j) vred[tid * 32 + j] = acc[j];
    __syncthreads();
    for (int st = 128; st > 0; st >>= 1) {
        if (tid < st)
            #pragma unroll
            for (int j = 0; j < 32; ++j)
                vred[tid * 32 + j] += vred[(tid + st) * 32 + j];
        __syncthreads();
    }
    if (tid < 32) osmall[(size_t)bh * 32 + tid] = vred[tid];
}

// ---------------------------------------------------------------------------
// Kernel 4: out_vec[b] = (osmall[b] @ Wproj + bproj) * layer_scale   (tiny)
// ---------------------------------------------------------------------------
__global__ __launch_bounds__(256) void proj_small(
    const float* __restrict__ osmall, const float* __restrict__ Wp,
    const float* __restrict__ bp, const float* __restrict__ ls,
    float* __restrict__ pbuf)
{
    const int b = blockIdx.x, ci = threadIdx.x;
    float s = 0.f;
    for (int j = 0; j < 256; ++j)
        s += osmall[(size_t)b * 256 + j] * Wp[(size_t)j * 256 + ci];
    pbuf[(size_t)b * 256 + ci] = (s + bp[ci]) * ls[ci];
}

// ---------------------------------------------------------------------------
// Kernel 5: broadcast per-batch 256-vector over all 4096 tokens (67 MB write)
// ---------------------------------------------------------------------------
__global__ __launch_bounds__(256) void bcast_out(
    const float* __restrict__ pbuf, float4* __restrict__ out)
{
    size_t idx = (size_t)blockIdx.x * 256 + threadIdx.x;   // over 16*4096*64
    int ci4 = (int)(idx & 63);
    size_t row = idx >> 6;                // b*4096 + pix
    int b = (int)(row >> 12);
    out[idx] = *(const float4*)(pbuf + (size_t)b * 256 + ci4 * 4);
}

// ---------------------------------------------------------------------------
extern "C" void kernel_launch(void* const* d_in, const int* in_sizes, int n_in,
                              void* d_out, int out_size, void* d_ws, size_t ws_size,
                              hipStream_t stream)
{
    const float* x_in   = (const float*)d_in[0];
    const float* Wq     = (const float*)d_in[1];
    const float* Wkv    = (const float*)d_in[2];
    const float* Wproj  = (const float*)d_in[3];
    const float* bproj  = (const float*)d_in[4];
    const float* conv_w = (const float*)d_in[5];
    const float* conv_b = (const float*)d_in[6];
    const float* lscale = (const float*)d_in[7];

    // workspace carve-up (~105 MB total)
    _Float16* qbuf = (_Float16*)d_ws;              // 16*8*4096*32 f16 = 32 MiB
    _Float16* kbuf = qbuf + (size_t)16777216;      // 32 MiB
    _Float16* vbuf = kbuf + (size_t)16777216;      // 32 MiB
    float* off     = (float*)(vbuf + (size_t)16777216);  // 16*16*4096 f32 = 4 MiB
    float* osmall  = off + (size_t)1048576;        // 16*8*32 f32
    float* pbuf    = osmall + 4096;                // 16*256 f32

    qkv_gemm   <<<dim3(512, 12), 256, 0, stream>>>(x_in, Wq, Wkv, qbuf, kbuf, vbuf);
    offset_conv<<<1024, 128, 0, stream>>>(x_in, conv_w, conv_b, off);
    attn_reduce<<<128, 256, 0, stream>>>(qbuf, kbuf, vbuf, off, osmall);
    proj_small <<<16, 256, 0, stream>>>(osmall, Wproj, bproj, lscale, pbuf);
    bcast_out  <<<16384, 256, 0, stream>>>(pbuf, (float4*)d_out);
}
